// NeuromorphicLM_36936718746187
// MI455X (gfx1250) — matrix operand order repaired and checked by reading, not compiled, not run on hardware
//
#include <hip/hip_runtime.h>
#include <math.h>

// ---------------- problem dims (match reference) ----------------
#define S_DIM   4
#define N_SEQ   1024
#define D_MOD   1024
#define B_BLK   8
#define C_COL   16
#define D_COL   32
#define G_COL   128
#define GD      4096        // G_COL * D_COL
#define RSLOTS  64
#define M_EM    256
#define C_EM    8
#define R_PASS  3
#define V_VOC   32000
#define SB_TOT  (S_DIM * B_BLK)                  // 32
#define ROWS_T  (S_DIM * N_SEQ * B_BLK * C_COL)  // 524288 column-rows
#define ETA     0.1f

typedef __attribute__((ext_vector_type(4)))  float  v4f;
typedef __attribute__((ext_vector_type(8)))  float  v8f;
typedef __attribute__((ext_vector_type(8)))  __bf16 v8bf;
typedef __attribute__((ext_vector_type(16))) __bf16 v16bf;

// ---------------- WMMA helpers ----------------
// A 16x32 bf16 fragment from row-major storage. Per ISA 7.12.2 the lane (m,hi)
// holds K = {8hi..8hi+7} and {16+8hi..23+8hi}: two contiguous 16B runs -> 2x b128.
__device__ __forceinline__ v16bf frag_a_rm(const __bf16* A, int lda, int lane) {
  int m = lane & 15, hi = lane >> 4;
  const __bf16* row = A + (size_t)m * lda + 8 * hi;
  v8bf lo = *(const v8bf*)(row);
  v8bf hh = *(const v8bf*)(row + 16);
  v16bf a;
#pragma unroll
  for (int j = 0; j < 8; ++j) { a[j] = lo[j]; a[8 + j] = hh[j]; }
  return a;
}
// B fragment from pre-swizzled storage: each 32x16 tile stored as 32 lanes x 16
// contiguous bf16 (32B per lane) -> single v16bf load (2x b128).
__device__ __forceinline__ v16bf frag_b_sw(const __bf16* Btile, int lane) {
  return *(const v16bf*)(Btile + lane * 16);
}
__device__ __forceinline__ v8f wmma_bf16(v16bf a, v16bf b, v8f c) {
  return __builtin_amdgcn_wmma_f32_16x16x32_bf16(false, a, false, b, (short)0, c,
                                                 false, false);
}
__device__ __forceinline__ float gelu_tanh(float x) {
  float x3 = x * x * x;
  return 0.5f * x * (1.0f + tanhf(0.7978845608028654f * (x + 0.044715f * x3)));
}

// ---------------- prep kernels ----------------
__global__ void embed_kernel(const int* ids, const float* emb, const float* pos,
                             __bf16* xb) {
  size_t i = (size_t)blockIdx.x * blockDim.x + threadIdx.x;  // over SN*D/8
  if (i >= (size_t)S_DIM * N_SEQ * (D_MOD / 8)) return;
  int sn = (int)(i / (D_MOD / 8));
  int dv = (int)(i % (D_MOD / 8)) * 8;
  int n = sn % N_SEQ;
  int tok = ids[sn];
  const float* e = emb + (size_t)tok * D_MOD + dv;
  const float* p = pos + (size_t)n * D_MOD + dv;
  v4f e0 = *(const v4f*)e, e1 = *(const v4f*)(e + 4);
  v4f p0 = *(const v4f*)p, p1 = *(const v4f*)(p + 4);
  v8bf o;
#pragma unroll
  for (int j = 0; j < 4; ++j) {
    o[j] = (__bf16)(e0[j] + p0[j]);
    o[4 + j] = (__bf16)(e1[j] + p1[j]);
  }
  *(v8bf*)(xb + i * 8) = o;
}

__global__ void cvt_f32_bf16_v8(const float* in, __bf16* out, size_t n8) {
  size_t i = (size_t)blockIdx.x * blockDim.x + threadIdx.x;
  if (i >= n8) return;
  const float* p = in + i * 8;
  v4f x0 = *(const v4f*)p, x1 = *(const v4f*)(p + 4);
  v8bf o;
#pragma unroll
  for (int j = 0; j < 4; ++j) { o[j] = (__bf16)x0[j]; o[4 + j] = (__bf16)x1[j]; }
  *(v8bf*)(out + i * 8) = o;
}

// Swizzle a [K][N] f32 matrix (optionally accessed transposed) into B-fragment
// order: tile (kt,nt) -> 512 contiguous bf16, lane L holds k=kt*32+8*(L>>4)+(j&7)+16*(j>>3),
// n=nt*16+(L&15). Output index == flat thread index by construction.
__global__ void swizzle_b_f32(const float* src, __bf16* dst, int K, int N,
                              int trans, int ldsrc) {
  size_t idx = (size_t)blockIdx.x * blockDim.x + threadIdx.x;
  if (idx >= (size_t)K * N) return;
  int j = (int)(idx & 15);
  int L = (int)((idx >> 4) & 31);
  size_t block = idx >> 9;
  int nblk = N >> 4;
  int nt = (int)(block % nblk), kt = (int)(block / nblk);
  int k = kt * 32 + ((L >> 4) << 3) + (j & 7) + ((j >> 3) << 4);
  int n = nt * 16 + (L & 15);
  float v = trans ? src[(size_t)n * ldsrc + k] : src[(size_t)k * ldsrc + n];
  dst[idx] = (__bf16)v;
}

__global__ void normalize_mem(const float* pmK, const float* emK, float* pmKn,
                              float* emKn) {
  int row = blockIdx.x * blockDim.x + threadIdx.x;
  const int pmRows = SB_TOT * RSLOTS;
  const int emRows = SB_TOT * M_EM;
  const float* src;
  float* dst;
  if (row < pmRows) {
    src = pmK + (size_t)row * D_COL;
    dst = pmKn + (size_t)row * D_COL;
  } else if (row < pmRows + emRows) {
    int r = row - pmRows;
    src = emK + (size_t)r * D_COL;
    dst = emKn + (size_t)r * D_COL;
  } else {
    return;
  }
  float s = 0.f;
#pragma unroll
  for (int d = 0; d < D_COL; ++d) s += src[d] * src[d];
  float inv = 1.0f / (sqrtf(s) + 1e-6f);
#pragma unroll
  for (int d = 0; d < D_COL; ++d) dst[d] = src[d] * inv;
}

// ------------- bf16 WMMA GEMM: C[M,N] = A[M,K] * Bsw[K,N] (+bias) -------------
// block = 128 threads (4 waves stacked in M over a shared 64-wide N range).
// Each wave owns 2 M-tiles x 4 N-tiles -> 8 v_wmma per 32-deep K step; this is
// the largest spill-free tile (64 acc VGPRs + ~48 fragment VGPRs). B-fragment
// addresses are identical across the 4 waves (cache broadcast reuse).
__global__ void __launch_bounds__(128)
gemm_bf16_wmma(const __bf16* __restrict__ A, const __bf16* __restrict__ Bsw,
               const float* __restrict__ bias, float* __restrict__ C, int M, int N,
               int K, int lda, int ldc) {
  int lane = threadIdx.x & 31;
  int wave = threadIdx.x >> 5;
  int nblk = N >> 4;
  int n0 = blockIdx.x * 64;
  int m0 = blockIdx.y * 128 + wave * 32;
  if (m0 >= M || n0 >= N) return;
  v8f acc[2][4] = {};
  const __bf16* A0 = A + (size_t)m0 * lda;
  const __bf16* A1 = A0 + (size_t)16 * lda;
  int nt0 = n0 >> 4;
  for (int kb = 0; kb < K; kb += 32) {
    const __bf16* Bp = Bsw + (((size_t)(kb >> 5) * nblk + nt0) << 9);
    if (kb + 32 < K) {  // gfx1250 global_prefetch_b8 on next K-step operands
      __builtin_prefetch(A0 + kb + 32, 0, 1);
      __builtin_prefetch(Bsw + (((size_t)((kb >> 5) + 1) * nblk + nt0) << 9) +
                             lane * 16, 0, 1);
    }
    v16bf a0 = frag_a_rm(A0 + kb, lda, lane);
    v16bf a1 = frag_a_rm(A1 + kb, lda, lane);
    v16bf b0 = frag_b_sw(Bp, lane);
    v16bf b1 = frag_b_sw(Bp + 512, lane);
    v16bf b2 = frag_b_sw(Bp + 1024, lane);
    v16bf b3 = frag_b_sw(Bp + 1536, lane);
    acc[0][0] = wmma_bf16(a0, b0, acc[0][0]);
    acc[0][1] = wmma_bf16(a0, b1, acc[0][1]);
    acc[0][2] = wmma_bf16(a0, b2, acc[0][2]);
    acc[0][3] = wmma_bf16(a0, b3, acc[0][3]);
    acc[1][0] = wmma_bf16(a1, b0, acc[1][0]);
    acc[1][1] = wmma_bf16(a1, b1, acc[1][1]);
    acc[1][2] = wmma_bf16(a1, b2, acc[1][2]);
    acc[1][3] = wmma_bf16(a1, b3, acc[1][3]);
  }
  int n = lane & 15, mb = (lane >> 4) * 8;
#pragma unroll
  for (int t = 0; t < 4; ++t) {
    int cc = n0 + t * 16 + n;
    float bb = bias ? bias[cc] : 0.f;
#pragma unroll
    for (int v = 0; v < 8; ++v) {
      C[(size_t)(m0 + mb + v) * ldc + cc] = acc[0][t][v] + bb;
      C[(size_t)(m0 + 16 + mb + v) * ldc + cc] = acc[1][t][v] + bb;
    }
  }
}

// ------------- per-column 32x32 chain: h=gelu(x@Wc+b), k=h@Wk, v=h@Wv ---------
__global__ void col_transform(const float* __restrict__ xcols,
                              const __bf16* __restrict__ wcS,
                              const float* __restrict__ bc,
                              const __bf16* __restrict__ wkS,
                              const __bf16* __restrict__ wvS,
                              __bf16* __restrict__ hbuf, __bf16* __restrict__ kbuf,
                              __bf16* __restrict__ vbuf) {
  __shared__ __bf16 sH[16 * 32];
  __shared__ __bf16 sK[16 * 32];
  __shared__ __bf16 sV[16 * 32];
  int lane = threadIdx.x;
  size_t row0 = (size_t)blockIdx.x * 16;

  int m = lane & 15, hi = lane >> 4;
  const float* arow = xcols + (row0 + m) * D_COL + 8 * hi;
  v4f a0 = *(const v4f*)(arow);
  v4f a1 = *(const v4f*)(arow + 4);
  v4f a2 = *(const v4f*)(arow + 16);
  v4f a3 = *(const v4f*)(arow + 20);
  v16bf a;
#pragma unroll
  for (int j = 0; j < 4; ++j) {
    a[j] = (__bf16)a0[j];
    a[4 + j] = (__bf16)a1[j];
    a[8 + j] = (__bf16)a2[j];
    a[12 + j] = (__bf16)a3[j];
  }

  v16bf w0 = frag_b_sw(wcS, lane);
  v16bf w1 = frag_b_sw(wcS + 512, lane);
  v8f h0 = {}, h1 = {};
  h0 = wmma_bf16(a, w0, h0);
  h1 = wmma_bf16(a, w1, h1);

  int n = lane & 15, mb = (lane >> 4) * 8;
#pragma unroll
  for (int v = 0; v < 8; ++v) {
    sH[(mb + v) * 32 + n] = (__bf16)gelu_tanh(h0[v] + bc[n]);
    sH[(mb + v) * 32 + 16 + n] = (__bf16)gelu_tanh(h1[v] + bc[16 + n]);
  }
  __syncthreads();

  v16bf ha = frag_a_rm(sH, 32, lane);
  v16bf bk0 = frag_b_sw(wkS, lane);
  v16bf bk1 = frag_b_sw(wkS + 512, lane);
  v16bf bv0 = frag_b_sw(wvS, lane);
  v16bf bv1 = frag_b_sw(wvS + 512, lane);
  v8f k0 = {}, k1 = {}, v0 = {}, v1 = {};
  k0 = wmma_bf16(ha, bk0, k0);
  k1 = wmma_bf16(ha, bk1, k1);
  v0 = wmma_bf16(ha, bv0, v0);
  v1 = wmma_bf16(ha, bv1, v1);

#pragma unroll
  for (int v = 0; v < 8; ++v) {
    sK[(mb + v) * 32 + n] = (__bf16)k0[v];
    sK[(mb + v) * 32 + 16 + n] = (__bf16)k1[v];
    sV[(mb + v) * 32 + n] = (__bf16)v0[v];
    sV[(mb + v) * 32 + 16 + n] = (__bf16)v1[v];
  }
  __syncthreads();

  int r = lane & 15, half = lane >> 4;
  size_t gof = (row0 + r) * D_COL + half * 16;
  *(v16bf*)(hbuf + gof) = *(const v16bf*)(sH + r * 32 + half * 16);
  *(v16bf*)(kbuf + gof) = *(const v16bf*)(sK + r * 32 + half * 16);
  *(v16bf*)(vbuf + gof) = *(const v16bf*)(sV + r * 32 + half * 16);
}

// ---------------- routing / memory read / eligibility / novelty ----------------
__device__ __forceinline__ void load_row32(const __bf16* p, float* dst) {
  const v8bf* q = (const v8bf*)p;
  v8bf r0 = q[0], r1 = q[1], r2 = q[2], r3 = q[3];
#pragma unroll
  for (int j = 0; j < 8; ++j) {
    dst[j] = (float)r0[j];
    dst[8 + j] = (float)r1[j];
    dst[16 + j] = (float)r2[j];
    dst[24 + j] = (float)r3[j];
  }
}

__global__ void routing_kernel(const __bf16* __restrict__ hbuf,
                               const __bf16* __restrict__ kbuf,
                               const __bf16* __restrict__ vbuf,
                               const float* __restrict__ pmKn,
                               const float* __restrict__ pmV,
                               const float* __restrict__ emKn,
                               const float* __restrict__ emV,
                               const float* __restrict__ wg,
                               const float* __restrict__ lamLogit,
                               float* __restrict__ xcols, float* __restrict__ nov,
                               float* __restrict__ eligK, float* __restrict__ eligV,
                               int first) {
  __shared__ float sPmKn[RSLOTS * D_COL];
  __shared__ float sPmV[RSLOTS * D_COL];
  __shared__ float sEK[RSLOTS * D_COL];
  __shared__ float sEV[RSLOTS * D_COL];
  __shared__ float sWg[D_COL];

  int tid = threadIdx.x;
  int sb = blockIdx.x >> 6;
  int chunk = blockIdx.x & 63;
  int s = sb / B_BLK, b = sb % B_BLK;
  size_t pmb = (size_t)sb * RSLOTS * D_COL;
  for (int i = tid; i < RSLOTS * D_COL; i += 256) {
    sPmKn[i] = pmKn[pmb + i];
    sPmV[i] = pmV[pmb + i];
    sEK[i] = 0.f;
    sEV[i] = 0.f;
  }
  if (tid < D_COL) sWg[tid] = wg[tid];
  __syncthreads();

  int rowlocal = chunk * 256 + tid;  // n*C + c
  int n = rowlocal / C_COL, c = rowlocal % C_COL;
  size_t rb = ((((size_t)s * N_SEQ + n) * B_BLK + b) * C_COL + c) * D_COL;

  float k[D_COL], vv[D_COL], xo[D_COL], acc[D_COL];
  load_row32(kbuf + rb, k);
  load_row32(vbuf + rb, vv);
  load_row32(hbuf + rb, xo);

  float nrm = 0.f, gdot = 0.f;
#pragma unroll
  for (int d = 0; d < D_COL; ++d) { nrm += k[d] * k[d]; gdot += xo[d] * sWg[d]; }
  float inv = 1.0f / (sqrtf(nrm) + 1e-6f);
  float gate = 1.0f / (1.0f + expf(-gdot));

  float pmmax = -1e30f;
  for (int q = 0; q < RSLOTS; ++q) {
    float sc = 0.f;
#pragma unroll
    for (int d = 0; d < D_COL; ++d) sc += k[d] * sPmKn[q * D_COL + d];
    pmmax = fmaxf(pmmax, sc * inv);
  }
  float pmsum = 0.f;
#pragma unroll
  for (int d = 0; d < D_COL; ++d) acc[d] = 0.f;
  for (int q = 0; q < RSLOTS; ++q) {
    float sc = 0.f;
#pragma unroll
    for (int d = 0; d < D_COL; ++d) sc += k[d] * sPmKn[q * D_COL + d];
    float w = expf(sc * inv - pmmax);
    pmsum += w;
#pragma unroll
    for (int d = 0; d < D_COL; ++d) acc[d] += w * sPmV[q * D_COL + d];
  }
  float rpm = 1.0f / pmsum;
#pragma unroll
  for (int d = 0; d < D_COL; ++d) xo[d] += acc[d] * rpm;
  for (int q = 0; q < RSLOTS; ++q) {
    float sc = 0.f;
#pragma unroll
    for (int d = 0; d < D_COL; ++d) sc += k[d] * sPmKn[q * D_COL + d];
    float g = gate * expf(sc * inv - pmmax) * rpm;
#pragma unroll
    for (int d = 0; d < D_COL; ++d) {
      atomicAdd(&sEK[q * D_COL + d], g * k[d]);
      atomicAdd(&sEV[q * D_COL + d], g * vv[d]);
    }
  }

  const float* eK = emKn + (size_t)sb * M_EM * D_COL;
  const float* eV = emV + (size_t)sb * M_EM * D_COL;
  float emmax = -1e30f;
  for (int m = 0; m < M_EM; ++m) {
    float sc = 0.f;
#pragma unroll
    for (int d = 0; d < D_COL; ++d) sc += k[d] * eK[m * D_COL + d];
    emmax = fmaxf(emmax, sc * inv);
  }
  float novv = gate * (1.0f - emmax);
  float emsum = 0.f;
#pragma unroll
  for (int d = 0; d < D_COL; ++d) acc[d] = 0.f;
  for (int m = 0; m < M_EM; ++m) {
    float sc = 0.f;
#pragma unroll
    for (int d = 0; d < D_COL; ++d) sc += k[d] * eK[m * D_COL + d];
    float w = expf(sc * inv - emmax);
    emsum += w;
#pragma unroll
    for (int d = 0; d < D_COL; ++d) acc[d] += w * eV[m * D_COL + d];
  }
  float rem = 1.0f / emsum;
#pragma unroll
  for (int d = 0; d < D_COL; ++d) xo[d] += acc[d] * rem;

  float lam = 1.0f / (1.0f + expf(-lamLogit[0]));
#pragma unroll
  for (int d = 0; d < D_COL; ++d) {
    size_t xi = rb + d;
    xcols[xi] = first ? xo[d] : (1.0f - lam) * xcols[xi] + lam * xo[d];
  }
  nov[(size_t)sb * (N_SEQ * C_COL) + rowlocal] = novv;

  __syncthreads();
  for (int i = tid; i < RSLOTS * D_COL; i += 256) {
    atomicAdd(&eligK[pmb + i], sEK[i]);
    atomicAdd(&eligV[pmb + i], sEV[i]);
  }
}

// ---------------- commit: pm update, novelty top-8, EM scatter ----------------
__global__ void commit_kernel(float* pmK, float* pmV, const float* eligK,
                              const float* eligV, float* emK, float* emV, float* emS,
                              const float* nov, const __bf16* kbuf,
                              const __bf16* vbuf) {
  int sb = blockIdx.x;
  int s = sb / B_BLK, b = sb % B_BLK;
  int tid = threadIdx.x;
  size_t pmb = (size_t)sb * RSLOTS * D_COL;
  for (int i = tid; i < RSLOTS * D_COL; i += 256) {
    pmK[pmb + i] += ETA * eligK[pmb + i];
    pmV[pmb + i] += ETA * eligV[pmb + i];
  }
  __shared__ float rv[256];
  __shared__ int ri[256];
  __shared__ int chosen[C_EM];
  __shared__ float chosenV[C_EM];
  __shared__ int low[C_EM];
  const float* novsb = nov + (size_t)sb * (N_SEQ * C_COL);

  for (int r = 0; r < C_EM; ++r) {
    float best = -1e30f;
    int bi = -1;
    for (int i = tid; i < N_SEQ * C_COL; i += 256) {
      bool skip = false;
      for (int j = 0; j < r; ++j)
        if (chosen[j] == i) skip = true;
      float v = novsb[i];
      if (!skip && (v > best || (v == best && (unsigned)i < (unsigned)bi))) {
        best = v;
        bi = i;
      }
    }
    rv[tid] = best;
    ri[tid] = bi;
    __syncthreads();
    for (int sft = 128; sft > 0; sft >>= 1) {
      if (tid < sft) {
        int o = tid + sft;
        if (rv[o] > rv[tid] ||
            (rv[o] == rv[tid] && (unsigned)ri[o] < (unsigned)ri[tid])) {
          rv[tid] = rv[o];
          ri[tid] = ri[o];
        }
      }
      __syncthreads();
    }
    if (tid == 0) { chosen[r] = ri[0]; chosenV[r] = rv[0]; }
    __syncthreads();
  }
  size_t emSb = (size_t)sb * M_EM;
  for (int r = 0; r < C_EM; ++r) {
    float best = 1e30f;
    int bi = -1;
    {
      bool skip = false;
      for (int j = 0; j < r; ++j)
        if (low[j] == tid) skip = true;
      if (!skip) { best = emS[emSb + tid]; bi = tid; }
    }
    rv[tid] = best;
    ri[tid] = bi;
    __syncthreads();
    for (int sft = 128; sft > 0; sft >>= 1) {
      if (tid < sft) {
        int o = tid + sft;
        if (rv[o] < rv[tid] ||
            (rv[o] == rv[tid] && (unsigned)ri[o] < (unsigned)ri[tid])) {
          rv[tid] = rv[o];
          ri[tid] = ri[o];
        }
      }
      __syncthreads();
    }
    if (tid == 0) low[r] = ri[0];
    __syncthreads();
  }
  int j = tid / D_COL, d = tid % D_COL;  // 256 = 8*32
  int src = chosen[j];
  int n = src / C_COL, c = src % C_COL;
  size_t rowb = ((((size_t)s * N_SEQ + n) * B_BLK + b) * C_COL + c) * D_COL;
  size_t dst = ((size_t)sb * M_EM + low[j]) * D_COL + d;
  emK[dst] = (float)kbuf[rowb + d];
  emV[dst] = (float)vbuf[rowb + d];
  if (d == 0) emS[emSb + low[j]] = chosenV[j];
}

// ---------------- layernorm (row per block) -> bf16 for LM head ---------------
__global__ void layernorm_kernel(const float* __restrict__ xr,
                                 const float* __restrict__ g,
                                 const float* __restrict__ bta,
                                 __bf16* __restrict__ xn) {
  __shared__ float s1[256], s2[256];
  int row = blockIdx.x, tid = threadIdx.x;
  const float* x = xr + (size_t)row * D_MOD;
  float a = 0.f, b = 0.f;
  for (int i = tid; i < D_MOD; i += 256) {
    float t = x[i];
    a += t;
    b += t * t;
  }
  s1[tid] = a;
  s2[tid] = b;
  __syncthreads();
  for (int sft = 128; sft > 0; sft >>= 1) {
    if (tid < sft) { s1[tid] += s1[tid + sft]; s2[tid] += s2[tid + sft]; }
    __syncthreads();
  }
  float mean = s1[0] / D_MOD;
  float var = s2[0] / D_MOD - mean * mean;
  float rstd = rsqrtf(var + 1e-5f);
  for (int i = tid; i < D_MOD; i += 256)
    xn[(size_t)row * D_MOD + i] = (__bf16)((x[i] - mean) * rstd * g[i] + bta[i]);
}

// ---------------- host orchestration ----------------
extern "C" void kernel_launch(void* const* d_in, const int* in_sizes, int n_in,
                              void* d_out, int out_size, void* d_ws, size_t ws_size,
                              hipStream_t stream) {
  (void)in_sizes; (void)n_in; (void)out_size; (void)ws_size;
  const int*   ids       = (const int*)d_in[0];
  const float* emb       = (const float*)d_in[1];
  const float* pos_emb   = (const float*)d_in[2];
  const float* W_fanout  = (const float*)d_in[3];
  const float* b_fanout  = (const float*)d_in[4];
  const float* W_col     = (const float*)d_in[5];
  const float* b_col     = (const float*)d_in[6];
  const float* Wk        = (const float*)d_in[7];
  const float* Wv        = (const float*)d_in[8];
  const float* wg        = (const float*)d_in[9];
  const float* pm_K      = (const float*)d_in[10];
  const float* pm_V      = (const float*)d_in[11];
  const float* em_K      = (const float*)d_in[12];
  const float* em_V      = (const float*)d_in[13];
  const float* em_S      = (const float*)d_in[14];
  const float* W_fanin   = (const float*)d_in[15];
  const float* b_fanin   = (const float*)d_in[16];
  const float* ln_g      = (const float*)d_in[17];
  const float* ln_b      = (const float*)d_in[18];
  const float* lam_logit = (const float*)d_in[19];
  float* out = (float*)d_out;

  size_t off = 0;
  auto alloc = [&](size_t bytes) -> void* {
    void* p = (char*)d_ws + off;
    off += (bytes + 255) & ~(size_t)255;
    return p;
  };
  const size_t SN = (size_t)S_DIM * N_SEQ;  // 4096
  const size_t XE = SN * GD;                // 16.7M column elems
  __bf16* xbf    = (__bf16*)alloc(SN * D_MOD * 2);
  __bf16* wfoS   = (__bf16*)alloc((size_t)D_MOD * GD * 2);      // swizzled
  __bf16* wfiS   = (__bf16*)alloc((size_t)GD * D_MOD * 2);      // swizzled
  __bf16* embS   = (__bf16*)alloc((size_t)D_MOD * V_VOC * 2);   // swizzled emb^T
  __bf16* wcS    = (__bf16*)alloc((size_t)D_COL * D_COL * 2);
  __bf16* wkS    = (__bf16*)alloc((size_t)D_COL * D_COL * 2);
  __bf16* wvS    = (__bf16*)alloc((size_t)D_COL * D_COL * 2);
  float*  xcols  = (float*)alloc(XE * 4);
  __bf16* xcolsB = (__bf16*)alloc(XE * 2);
  __bf16* hb     = (__bf16*)alloc(XE * 2);
  __bf16* kb     = (__bf16*)alloc(XE * 2);
  __bf16* vb     = (__bf16*)alloc(XE * 2);
  float*  nov    = (float*)alloc((size_t)SB_TOT * N_SEQ * C_COL * 4);
  float*  pmKc   = (float*)alloc((size_t)SB_TOT * RSLOTS * D_COL * 4);
  float*  pmVc   = (float*)alloc((size_t)SB_TOT * RSLOTS * D_COL * 4);
  float*  emKc   = (float*)alloc((size_t)SB_TOT * M_EM * D_COL * 4);
  float*  emVc   = (float*)alloc((size_t)SB_TOT * M_EM * D_COL * 4);
  float*  emSc   = (float*)alloc((size_t)SB_TOT * M_EM * 4);
  float*  pmKn   = (float*)alloc((size_t)SB_TOT * RSLOTS * D_COL * 4);
  float*  emKn   = (float*)alloc((size_t)SB_TOT * M_EM * D_COL * 4);
  float*  eligK  = (float*)alloc((size_t)SB_TOT * RSLOTS * D_COL * 4);
  float*  eligV  = (float*)alloc((size_t)SB_TOT * RSLOTS * D_COL * 4);
  float*  xr     = (float*)alloc(SN * D_MOD * 4);
  __bf16* xnB    = (__bf16*)alloc(SN * D_MOD * 2);

  hipMemcpyAsync(pmKc, pm_K, (size_t)SB_TOT * RSLOTS * D_COL * 4, hipMemcpyDeviceToDevice, stream);
  hipMemcpyAsync(pmVc, pm_V, (size_t)SB_TOT * RSLOTS * D_COL * 4, hipMemcpyDeviceToDevice, stream);
  hipMemcpyAsync(emKc, em_K, (size_t)SB_TOT * M_EM * D_COL * 4, hipMemcpyDeviceToDevice, stream);
  hipMemcpyAsync(emVc, em_V, (size_t)SB_TOT * M_EM * D_COL * 4, hipMemcpyDeviceToDevice, stream);
  hipMemcpyAsync(emSc, em_S, (size_t)SB_TOT * M_EM * 4, hipMemcpyDeviceToDevice, stream);

  embed_kernel<<<(SN * D_MOD / 8 + 255) / 256, 256, 0, stream>>>(ids, emb, pos_emb, xbf);
  swizzle_b_f32<<<((size_t)D_MOD * GD + 255) / 256, 256, 0, stream>>>(
      W_fanout, wfoS, D_MOD, GD, 0, GD);
  swizzle_b_f32<<<((size_t)GD * D_MOD + 255) / 256, 256, 0, stream>>>(
      W_fanin, wfiS, GD, D_MOD, 0, D_MOD);
  swizzle_b_f32<<<((size_t)D_MOD * V_VOC + 255) / 256, 256, 0, stream>>>(
      emb, embS, D_MOD, V_VOC, 1, D_MOD);
  swizzle_b_f32<<<4, 256, 0, stream>>>(W_col, wcS, D_COL, D_COL, 0, D_COL);
  swizzle_b_f32<<<4, 256, 0, stream>>>(Wk, wkS, D_COL, D_COL, 0, D_COL);
  swizzle_b_f32<<<4, 256, 0, stream>>>(Wv, wvS, D_COL, D_COL, 0, D_COL);

  // fanout GEMM: [4096,1024] x [1024,4096] -> x_cols
  gemm_bf16_wmma<<<dim3(GD / 64, SN / 128), 128, 0, stream>>>(
      xbf, wfoS, b_fanout, xcols, (int)SN, GD, D_MOD, D_MOD, GD);

  // refinement passes
  for (int r = 0; r < R_PASS; ++r) {
    normalize_mem<<<(SB_TOT * (RSLOTS + M_EM) + 255) / 256, 256, 0, stream>>>(
        pmKc, emKc, pmKn, emKn);
    col_transform<<<ROWS_T / 16, 32, 0, stream>>>(xcols, wcS, b_col, wkS, wvS, hb, kb, vb);
    hipMemsetAsync(eligK, 0, (size_t)SB_TOT * RSLOTS * D_COL * 4, stream);
    hipMemsetAsync(eligV, 0, (size_t)SB_TOT * RSLOTS * D_COL * 4, stream);
    routing_kernel<<<SB_TOT * 64, 256, 0, stream>>>(
        hb, kb, vb, pmKn, pmVc, emKn, emVc, wg, lam_logit, xcols, nov, eligK, eligV,
        r == 0 ? 1 : 0);
    commit_kernel<<<SB_TOT, 256, 0, stream>>>(pmKc, pmVc, eligK, eligV, emKc, emVc,
                                              emSc, nov, kb, vb);
  }

  // fanin GEMM: [4096,4096] x [4096,1024] -> xr
  cvt_f32_bf16_v8<<<(XE / 8 + 255) / 256, 256, 0, stream>>>(xcols, xcolsB, XE / 8);
  gemm_bf16_wmma<<<dim3(D_MOD / 64, SN / 128), 128, 0, stream>>>(
      xcolsB, wfiS, b_fanin, xr, (int)SN, D_MOD, GD, GD, D_MOD);

  // layernorm + tied LM head: [4096,1024] x [1024,32000] -> logits
  layernorm_kernel<<<(int)SN, 256, 0, stream>>>(xr, ln_g, ln_b, xnB);
  gemm_bf16_wmma<<<dim3(V_VOC / 64, SN / 128), 128, 0, stream>>>(
      xnB, embS, nullptr, out, (int)SN, V_VOC, D_MOD, D_MOD, V_VOC);
}
